// TridentTransformer_64115271795078
// MI455X (gfx1250) — matrix operand order repaired and checked by reading, not compiled, not run on hardware
//
#include <hip/hip_runtime.h>
#include <math.h>

typedef __bf16 bf16_t;
typedef __bf16 v16bf __attribute__((ext_vector_type(16)));
typedef float  v8f   __attribute__((ext_vector_type(8)));
typedef unsigned int u32x4 __attribute__((ext_vector_type(4)));
typedef int i32x4 __attribute__((ext_vector_type(4)));
typedef int i32x8 __attribute__((ext_vector_type(8)));

#define TPB 256

#if defined(__has_builtin)
#if __has_builtin(__builtin_amdgcn_tensor_load_to_lds)
#define HAVE_TDM 1
#endif
#endif

__device__ __forceinline__ float gelu_f(float x) {
    return 0.5f * x * (1.0f + erff(x * 0.70710678118654752440f));
}
__device__ __forceinline__ float sigm_f(float x) {
    return 1.0f / (1.0f + __expf(-x));
}

#ifdef HAVE_TDM
// ---------------------------------------------------------------------------
// Tensor Data Mover: DMA a rows x cols tile of 2-byte elements from global
// (row stride = rowStrideElems elements) into LDS at byte offset ldsOff.
// Descriptor layout per CDNA5 ISA.
// ---------------------------------------------------------------------------
__device__ __forceinline__ void tdm_load_2d(unsigned ldsOff, const void* gptr,
                                            unsigned rows, unsigned cols,
                                            unsigned rowStrideElems,
                                            unsigned tensorW, unsigned tensorH) {
    unsigned long long ga = (unsigned long long)gptr;
    u32x4 g0;
    g0[0] = 1u;                                            // count=1, user desc
    g0[1] = ldsOff;                                        // lds_addr (bytes)
    g0[2] = (unsigned)ga;                                  // global_addr[31:0]
    g0[3] = (unsigned)((ga >> 32) & 0x01ffffffu) | (2u << 30);  // addr hi | type=2
    i32x8 g1;
    g1[0] = 0x00010000;                                    // data_size = 1 (2B)
    g1[1] = (int)((tensorW & 0xffffu) << 16);              // tensor_dim0[15:0]
    g1[2] = (int)((tensorW >> 16) | ((tensorH & 0xffffu) << 16));  // dim0 hi|dim1 lo
    g1[3] = (int)((tensorH >> 16) | (cols << 16));         // dim1 hi | tile_dim0
    g1[4] = (int)(rows & 0xffffu);                         // tile_dim1, tile_dim2=0
    g1[5] = (int)rowStrideElems;                           // tensor_dim0_stride lo
    g1[6] = 0;
    g1[7] = 0;
    i32x4 gz;
    gz[0] = 0; gz[1] = 0; gz[2] = 0; gz[3] = 0;
#if __clang_major__ >= 23
    i32x8 gz8;
    gz8[0] = 0; gz8[1] = 0; gz8[2] = 0; gz8[3] = 0;
    gz8[4] = 0; gz8[5] = 0; gz8[6] = 0; gz8[7] = 0;
    __builtin_amdgcn_tensor_load_to_lds(g0, g1, gz, gz, gz8, 0);
#else
    __builtin_amdgcn_tensor_load_to_lds(g0, g1, gz, gz, 0);
#endif
}
#endif

// ---------------------------------------------------------------------------
// Channel LayerNorm over C channels at each (b, s) pixel. Layout (B, C, S).
// ---------------------------------------------------------------------------
__global__ void ln_ch_kernel(const float* __restrict__ x, const float* __restrict__ w,
                             const float* __restrict__ b, float* __restrict__ y,
                             int C, long S, long total) {
    long i = (long)blockIdx.x * blockDim.x + threadIdx.x;
    if (i >= total) return;
    long bb = i / S, s = i % S;
    const float* xp = x + bb * (long)C * S + s;
    float mu = 0.f;
    for (int c = 0; c < C; ++c) mu += xp[(long)c * S];
    mu /= (float)C;
    float var = 0.f;
    for (int c = 0; c < C; ++c) { float d = xp[(long)c * S] - mu; var += d * d; }
    var /= (float)C;
    float inv = rsqrtf(var + 1e-5f);
    float* yp = y + bb * (long)C * S + s;
    for (int c = 0; c < C; ++c)
        yp[(long)c * S] = (xp[(long)c * S] - mu) * inv * w[c] + b[c];
}

// ---------------------------------------------------------------------------
// f32 -> bf16 conversion with independent batch strides (for concat targets).
// ---------------------------------------------------------------------------
__global__ void cvt_bf16(const float* __restrict__ s, long sBS, bf16_t* __restrict__ d,
                         long dBS, long per, long total) {
    long i = (long)blockIdx.x * blockDim.x + threadIdx.x;
    if (i >= total) return;
    long bb = i / per, r = i % per;
    d[bb * dBS + r] = (bf16_t)s[bb * sBS + r];
}

// ---------------------------------------------------------------------------
// WMMA bf16 GEMM: C[b][m][n] = act(sum_k A[m,k]*B[b](k,n) + bias[m]) (+res)
// A: (M,K) bf16 row-major weights. B element (k,n) at Bp[k*ldbK + n*ldbN].
// grid = (N/64, M/16, batches), block = 128 (4 waves, 1 n-tile each).
// TDMP: stage B tiles through LDS with the Tensor Data Mover, double-buffered
// per wave (each wave owns its LDS slot -> no cross-wave barriers; TDM ops of
// one wave complete in order, so s_wait_tensorcnt(1) means the older tile is
// resident).  Requires ldbN == 1.
// ---------------------------------------------------------------------------
template<int ACT, bool HAS_BIAS, bool HAS_RES, bool TDMP>
__global__ void wmma_gemm(const bf16_t* __restrict__ A, const bf16_t* __restrict__ Bp,
                          long bBatch, const float* __restrict__ bias,
                          const float* __restrict__ res, long rBatch,
                          float* __restrict__ C, long cBatch,
                          int M, int K, int N, long ldbK, long ldbN) {
#ifdef HAVE_TDM
    constexpr bool useTdm = TDMP;
#else
    constexpr bool useTdm = false;
#endif
    __shared__ bf16_t Bt[2][4][32 * 16];  // [buf][wave][k*16 + n]
    int lane = threadIdx.x & 31;
    int wave = threadIdx.x >> 5;
    int n0 = (blockIdx.x * 4 + wave) * 16;
    int m0 = blockIdx.y * 16;
    int b = blockIdx.z;
    if (n0 >= N) return;  // uniform per-wave; never taken for our grids
    const bf16_t* Bb = Bp + (long)b * bBatch;
    float* Cb = C + (long)b * cBatch;

    int am = m0 + (lane & 15);
    int akb = (lane >> 4) * 16;      // A: K 0-15 (lo lanes) vs 16-31
    int bn = n0 + (lane & 15);
    int bkb = (lane >> 4) * 16;      // B: K 0-15 (lo lanes) vs 16-31
    int nsteps = K >> 5;

#ifdef HAVE_TDM
    if (useTdm) {
        tdm_load_2d((unsigned)(unsigned long long)&Bt[0][wave][0],
                    Bb + (long)n0, 32, 16, (unsigned)ldbK, (unsigned)N, (unsigned)K);
    }
#endif
    v8f acc = {};
    for (int s = 0; s < nsteps; ++s) {
        int k0 = s << 5;
        const bf16_t* Arow = A + (long)am * K + k0;
        __builtin_prefetch(Arow + 32, 0, 1);
        v16bf af, bfv;
#pragma unroll
        for (int e = 0; e < 16; ++e) {
            int vg = e >> 1, hh = e & 1;
            int ka = akb + e;
            af[e] = Arow[ka];
        }
        if (useTdm) {
#ifdef HAVE_TDM
            if (s + 1 < nsteps) {
                tdm_load_2d((unsigned)(unsigned long long)&Bt[(s + 1) & 1][wave][0],
                            Bb + (long)(k0 + 32) * ldbK + n0, 32, 16,
                            (unsigned)ldbK, (unsigned)N, (unsigned)K);
                __builtin_amdgcn_s_wait_tensorcnt(1);
            } else {
                __builtin_amdgcn_s_wait_tensorcnt(0);
            }
            const bf16_t* bt = &Bt[s & 1][wave][0];
#pragma unroll
            for (int e = 0; e < 16; ++e)
                bfv[e] = bt[(bkb + e) * 16 + (lane & 15)];
#endif
        } else {
#pragma unroll
            for (int e = 0; e < 16; ++e) {
                long kb = k0 + bkb + e;
                bfv[e] = Bb[kb * ldbK + (long)bn * ldbN];
            }
        }
        acc = __builtin_amdgcn_wmma_f32_16x16x32_bf16(false, af, false, bfv,
                                                      (short)0, acc, false, false);
    }
    // D layout: VGPR i, lanes 0-15 -> (m=i, n=lane); lanes 16-31 -> (m=i+8)
    int cn = n0 + (lane & 15);
    int mbase = m0 + ((lane >> 4) << 3);
#pragma unroll
    for (int i = 0; i < 8; ++i) {
        int cm = mbase + i;
        float val = acc[i];
        if (HAS_BIAS) val += bias[cm];
        if (ACT == 1) val = gelu_f(val);
        if (HAS_RES) val += res[(long)b * rBatch + (long)cm * N + cn];
        Cb[(long)cm * N + cn] = val;
    }
}

// ---------------------------------------------------------------------------
// Depthwise 3x3, stride 1, pad 1, optional bias / gelu. Layout (B,C,H,W).
// ---------------------------------------------------------------------------
template<bool GELU>
__global__ void dw3x3(const float* __restrict__ x, const float* __restrict__ w,
                      const float* __restrict__ bias, float* __restrict__ y,
                      int C, int H, int W, long total) {
    long i = (long)blockIdx.x * blockDim.x + threadIdx.x;
    if (i >= total) return;
    long HW = (long)H * W;
    int s = (int)(i % HW);
    long bc = i / HW;
    int c = (int)(bc % C);
    int h = s / W, wv = s % W;
    const float* xp = x + bc * HW;
    const float* wp = w + c * 9;
    float acc = bias ? bias[c] : 0.f;
    for (int kh = 0; kh < 3; ++kh) {
        int ih = h + kh - 1;
        if (ih < 0 || ih >= H) continue;
        for (int kw = 0; kw < 3; ++kw) {
            int iw = wv + kw - 1;
            if (iw < 0 || iw >= W) continue;
            acc += wp[kh * 3 + kw] * xp[(long)ih * W + iw];
        }
    }
    y[i] = GELU ? gelu_f(acc) : acc;
}

// Depthwise 2x2, stride 2, pad 0, bias. out: (B,C,H/2,W/2)
__global__ void dw2x2s2(const float* __restrict__ x, const float* __restrict__ w,
                        const float* __restrict__ bias, float* __restrict__ y,
                        int C, int H, int W, long total) {
    long i = (long)blockIdx.x * blockDim.x + threadIdx.x;
    if (i >= total) return;
    int H2 = H >> 1, W2 = W >> 1;
    long HW2 = (long)H2 * W2;
    int s = (int)(i % HW2);
    long bc = i / HW2;
    int c = (int)(bc % C);
    int h2 = s / W2, w2 = s % W2;
    const float* xp = x + bc * (long)H * W;
    const float* wp = w + c * 4;
    long base = (long)(2 * h2) * W + 2 * w2;
    float acc = bias[c] + wp[0] * xp[base] + wp[1] * xp[base + 1] +
                wp[2] * xp[base + W] + wp[3] * xp[base + W + 1];
    y[i] = acc;
}

// Grouped conv (2 in -> 2 out per group), 3x3 pad 1, no bias (qkvdw).
__global__ void grp2_3x3(const float* __restrict__ x, const float* __restrict__ w,
                         float* __restrict__ y, int Cout, int H, int W, long total) {
    long i = (long)blockIdx.x * blockDim.x + threadIdx.x;
    if (i >= total) return;
    long HW = (long)H * W;
    int s = (int)(i % HW);
    long bco = i / HW;
    int co = (int)(bco % Cout);
    long bb = bco / Cout;
    int g = co >> 1;
    int h = s / W, wv = s % W;
    const float* x0 = x + (bb * Cout + 2 * g) * HW;
    const float* x1 = x0 + HW;
    const float* wp = w + (long)co * 18;
    float acc = 0.f;
    for (int kh = 0; kh < 3; ++kh) {
        int ih = h + kh - 1;
        if (ih < 0 || ih >= H) continue;
        for (int kw = 0; kw < 3; ++kw) {
            int iw = wv + kw - 1;
            if (iw < 0 || iw >= W) continue;
            long o = (long)ih * W + iw;
            acc += wp[kh * 3 + kw] * x0[o] + wp[9 + kh * 3 + kw] * x1[o];
        }
    }
    y[i] = acc;
}

// Grouped conv (2 in -> 2 out per group), 3x5 pad (1,2), no bias (v2).
__global__ void grp2_3x5(const float* __restrict__ x, const float* __restrict__ w,
                         float* __restrict__ y, int Cout, int H, int W, long total) {
    long i = (long)blockIdx.x * blockDim.x + threadIdx.x;
    if (i >= total) return;
    long HW = (long)H * W;
    int s = (int)(i % HW);
    long bco = i / HW;
    int co = (int)(bco % Cout);
    long bb = bco / Cout;
    int g = co >> 1;
    int h = s / W, wv = s % W;
    const float* x0 = x + (bb * Cout + 2 * g) * HW;
    const float* x1 = x0 + HW;
    const float* wp = w + (long)co * 30;
    float acc = 0.f;
    for (int kh = 0; kh < 3; ++kh) {
        int ih = h + kh - 1;
        if (ih < 0 || ih >= H) continue;
        for (int kw = 0; kw < 5; ++kw) {
            int iw = wv + kw - 2;
            if (iw < 0 || iw >= W) continue;
            long o = (long)ih * W + iw;
            acc += wp[kh * 5 + kw] * x0[o] + wp[15 + kh * 5 + kw] * x1[o];
        }
    }
    y[i] = acc;
}

// Grouped 1x1 (1 in -> 2 out per group), bias (v1). x: (B,64,S) -> y: (B,128,S)
__global__ void v1_kernel(const float* __restrict__ x, const float* __restrict__ w,
                          const float* __restrict__ b, float* __restrict__ y,
                          long S, long total) {
    long i = (long)blockIdx.x * blockDim.x + threadIdx.x;
    if (i >= total) return;
    long s = i % S;
    long bco = i / S;
    int co = (int)(bco % 128);
    long bb = bco / 128;
    int ci = co >> 1;
    y[i] = x[(bb * 64 + ci) * S + s] * w[co] + b[co];
}

// Depthwise transposed conv 4x4 stride 2 pad 1: (B,C,Hi,Wi) -> (B,C,2Hi,2Wi)
__global__ void dwT4x4s2(const float* __restrict__ x, const float* __restrict__ w,
                         const float* __restrict__ bias, float* __restrict__ y,
                         int C, int Hi, int Wi, long total) {
    long i = (long)blockIdx.x * blockDim.x + threadIdx.x;
    if (i >= total) return;
    int Ho = 2 * Hi, Wo = 2 * Wi;
    long HWo = (long)Ho * Wo;
    int s = (int)(i % HWo);
    long bc = i / HWo;
    int c = (int)(bc % C);
    int ho = s / Wo, wo = s % Wo;
    const float* xp = x + bc * (long)Hi * Wi;
    const float* wp = w + (long)c * 16;
    float acc = bias[c];
    for (int kh = 0; kh < 4; ++kh) {
        int d = ho + kh - 2;
        if (d < 0 || d > 2 * Hi - 2 || (d & 1)) continue;
        int ih = d >> 1;
        for (int kw = 0; kw < 4; ++kw) {
            int e = wo + kw - 2;
            if (e < 0 || e > 2 * Wi - 2 || (e & 1)) continue;
            int iw = e >> 1;
            acc += wp[(3 - kh) * 4 + (3 - kw)] * xp[(long)ih * Wi + iw];
        }
    }
    y[i] = acc;
}

// Per-(b,c) spatial mean. One block per (b,c).
__global__ void chan_mean(const float* __restrict__ x, float* __restrict__ out, long S) {
    long bc = blockIdx.x;
    const float* p = x + bc * S;
    float a = 0.f;
    for (long i = threadIdx.x; i < S; i += blockDim.x) a += p[i];
    __shared__ float red[256];
    red[threadIdx.x] = a;
    __syncthreads();
    for (int st = 128; st > 0; st >>= 1) {
        if ((int)threadIdx.x < st) red[threadIdx.x] += red[threadIdx.x + st];
        __syncthreads();
    }
    if (threadIdx.x == 0) out[bc] = red[0] / (float)S;
}

// SE gate: per batch, 256 threads.
__global__ void se_gate(const float* __restrict__ mean, const float* __restrict__ w1,
                        const float* __restrict__ b1, const float* __restrict__ w2,
                        const float* __restrict__ b2, float* __restrict__ se) {
    int b = blockIdx.x, t = threadIdx.x;
    __shared__ float m[256];
    __shared__ float hbuf[64];
    m[t] = mean[b * 256 + t];
    __syncthreads();
    if (t < 64) {
        float a = b1[t];
        for (int c = 0; c < 256; ++c) a += w1[t * 256 + c] * m[c];
        hbuf[t] = gelu_f(a);
    }
    __syncthreads();
    float a = b2[t];
    for (int j = 0; j < 64; ++j) a += w2[t * 64 + j] * hbuf[j];
    se[b * 256 + t] = sigm_f(a);
}

// gi gate: per batch, 128 threads. gi = sigmoid(W2 * gelu(LN8(W1 * mean64)))
__global__ void gi_gate(const float* __restrict__ mean64, const float* __restrict__ w1,
                        const float* __restrict__ lnw, const float* __restrict__ lnb,
                        const float* __restrict__ w2, float* __restrict__ gi) {
    int b = blockIdx.x, t = threadIdx.x;
    __shared__ float m[64];
    __shared__ float g1[8];
    __shared__ float g1n[8];
    if (t < 64) m[t] = mean64[b * 64 + t];
    __syncthreads();
    if (t < 8) {
        float a = 0.f;
        for (int c = 0; c < 64; ++c) a += w1[t * 64 + c] * m[c];
        g1[t] = a;
    }
    __syncthreads();
    if (t == 0) {
        float mu = 0.f;
        for (int j = 0; j < 8; ++j) mu += g1[j];
        mu *= 0.125f;
        float var = 0.f;
        for (int j = 0; j < 8; ++j) { float d = g1[j] - mu; var += d * d; }
        var *= 0.125f;
        float inv = rsqrtf(var + 1e-5f);
        for (int j = 0; j < 8; ++j)
            g1n[j] = gelu_f((g1[j] - mu) * inv * lnw[j] + lnb[j]);
    }
    __syncthreads();
    float a = 0.f;
    for (int j = 0; j < 8; ++j) a += w2[t * 8 + j] * g1n[j];
    gi[b * 128 + t] = sigm_f(a);
}

// y_bf16[b,c,s] = bf16(x[b,c,s] * scale[b*C+c])
__global__ void scale_ch_bf16(const float* __restrict__ x, const float* __restrict__ scale,
                              bf16_t* __restrict__ y, int C, long S, long total) {
    long i = (long)blockIdx.x * blockDim.x + threadIdx.x;
    if (i >= total) return;
    long bc = i / S;
    int c = (int)(bc % C);
    long b = bc / C;
    y[i] = (bf16_t)(x[i] * scale[b * C + c]);
}

// y[b,s] = sum_c x[b,c,s] * w[c]  (spatial interaction dot)
__global__ void dot_ch(const float* __restrict__ x, const float* __restrict__ w,
                       float* __restrict__ y, int C, long S, long total) {
    long i = (long)blockIdx.x * blockDim.x + threadIdx.x;
    if (i >= total) return;
    long b = i / S, s = i % S;
    const float* xp = x + b * (long)C * S + s;
    float a = 0.f;
    for (int c = 0; c < C; ++c) a += xp[(long)c * S] * w[c];
    y[i] = a;
}

__global__ void gelu_ip(float* __restrict__ x, long total) {
    long i = (long)blockIdx.x * blockDim.x + threadIdx.x;
    if (i >= total) return;
    x[i] = gelu_f(x[i]);
}

// x[b,c,s] *= m[b,s]
__global__ void mul_bcast(float* __restrict__ x, const float* __restrict__ m,
                          int C, long S, long total) {
    long i = (long)blockIdx.x * blockDim.x + threadIdx.x;
    if (i >= total) return;
    long s = i % S;
    long b = (i / S) / C;
    x[i] *= m[b * S + s];
}

// d = bf16(a + b + c)
__global__ void add3_bf16(const float* __restrict__ a, const float* __restrict__ b,
                          const float* __restrict__ c, bf16_t* __restrict__ d, long total) {
    long i = (long)blockIdx.x * blockDim.x + threadIdx.x;
    if (i >= total) return;
    d[i] = (bf16_t)(a[i] + b[i] + c[i]);
}

// FFN gate fuse: d = bf16(gelu(t2[:, :128]) * t2[:, 128:])
__global__ void gate_fuse(const float* __restrict__ t2, bf16_t* __restrict__ d, long total) {
    long i = (long)blockIdx.x * blockDim.x + threadIdx.x;
    if (i >= total) return;
    const long per = 128L * 65536L;
    long bb = i / per, r = i % per;
    const float* base = t2 + bb * 256L * 65536L;
    d[i] = (bf16_t)(gelu_f(base[r]) * base[r + per]);
}

// kz L2-normalize over head-dim (16) per (b, h, token).
__global__ void kz_norm(const float* __restrict__ kv, float* __restrict__ kzn, long total) {
    long i = (long)blockIdx.x * blockDim.x + threadIdx.x;
    if (i >= total) return;
    int tok = (int)(i & 255);
    int h = (int)((i >> 8) & 7);
    int b = (int)(i >> 11);
    const float* base = kv + ((long)b * 256 + h * 16) * 256 + tok;
    float ss = 0.f;
    for (int c = 0; c < 16; ++c) { float v = base[(long)c * 256]; ss += v * v; }
    float inv = 1.0f / fmaxf(sqrtf(ss), 1e-12f);
    float* dst = kzn + ((long)b * 128 + h * 16) * 256 + tok;
    for (int c = 0; c < 16; ++c) dst[(long)c * 256] = base[(long)c * 256] * inv;
}

// ---------------------------------------------------------------------------
// Spectral (channel) attention: per (b,h) gram(16x16 over n=16384) via WMMA,
// softmax, apply to v. Block = 256 threads (8 waves), grid = B*8.
// ---------------------------------------------------------------------------
__global__ void attn1_kernel(const float* __restrict__ qk, const float* __restrict__ v,
                             const float* __restrict__ temp, float* __restrict__ o) {
    const int Sh = 16384;
    int b = blockIdx.x >> 3;
    int h = blockIdx.x & 7;
    int tid = threadIdx.x;
    int lane = tid & 31, wave = tid >> 5;
    __shared__ float inv[32];
    __shared__ float red[32][8];
    __shared__ float part[8][32][8];
    __shared__ float Wm[16][16];

    const float* qb = qk + ((long)b * 256 + h * 16) * Sh;
    const float* kb = qk + ((long)b * 256 + 128 + h * 16) * Sh;

    // phase 1: channel L2 norms (16 q + 16 k)
    {
        int ch = tid >> 3, p = tid & 7;
        const float* src = (ch < 16) ? (qb + (long)ch * Sh) : (kb + (long)(ch - 16) * Sh);
        float ss = 0.f;
        for (int n = p; n < Sh; n += 8) { float t = src[n]; ss += t * t; }
        red[ch][p] = ss;
    }
    __syncthreads();
    if (tid < 32) {
        float ss = 0.f;
        for (int j = 0; j < 8; ++j) ss += red[tid][j];
        inv[tid] = 1.0f / fmaxf(sqrtf(ss), 1e-12f);
    }
    __syncthreads();
    // phase 2: gram via bf16 WMMA, each wave covers a 2048-wide n chunk
    {
        int am = lane & 15;
        int akb = (lane >> 4) * 16;
        int bn = lane & 15;
        int bkb = (lane >> 4) * 16;
        float qs = inv[am];
        float ks = inv[16 + bn];
        const float* qrow = qb + (long)am * Sh;
        const float* krow = kb + (long)bn * Sh;
        v8f acc = {};
        for (int n0 = wave * 2048; n0 < (wave + 1) * 2048; n0 += 32) {
            v16bf af, bfv;
#pragma unroll
            for (int e = 0; e < 16; ++e) {
                int vg = e >> 1, hh = e & 1;
                int ka = akb + e;
                af[e] = (bf16_t)(qrow[n0 + ka] * qs);
            }
#pragma unroll
            for (int e = 0; e < 16; ++e) bfv[e] = (bf16_t)(krow[n0 + bkb + e] * ks);
            acc = __builtin_amdgcn_wmma_f32_16x16x32_bf16(false, af, false, bfv,
                                                          (short)0, acc, false, false);
        }
#pragma unroll
        for (int i = 0; i < 8; ++i) part[wave][lane][i] = acc[i];
    }
    __syncthreads();
    // phase 3: reduce partial grams, scale by temperature
    {
        int m = tid >> 4, d = tid & 15;
        int ln = d + ((m >= 8) ? 16 : 0);
        int vg = m & 7;
        float sum = 0.f;
        for (int wv = 0; wv < 8; ++wv) sum += part[wv][ln][vg];
        Wm[m][d] = sum * temp[h];
    }
    __syncthreads();
    // phase 4: softmax rows
    if (tid < 16) {
        float mx = -3.4e38f;
        for (int d = 0; d < 16; ++d) mx = fmaxf(mx, Wm[tid][d]);
        float s = 0.f;
        for (int d = 0; d < 16; ++d) { float e = __expf(Wm[tid][d] - mx); Wm[tid][d] = e; s += e; }
        float is = 1.0f / s;
        for (int d = 0; d < 16; ++d) Wm[tid][d] *= is;
    }
    __syncthreads();
    // phase 5: o[c,n] = sum_d Wm[c][d] * v[d][n]
    const float* vb = v + ((long)b * 128 + h * 16) * Sh;
    float* ob = o + ((long)b * 128 + h * 16) * Sh;
    for (int n = tid; n < Sh; n += 256) {
        float vv[16];
#pragma unroll
        for (int d = 0; d < 16; ++d) vv[d] = vb[(long)d * Sh + n];
#pragma unroll
        for (int c = 0; c < 16; ++c) {
            float a = 0.f;
#pragma unroll
            for (int d = 0; d < 16; ++d) a += Wm[c][d] * vv[d];
            ob[(long)c * Sh + n] = a;
        }
    }
}

// ---------------------------------------------------------------------------
// Prior cross-attention: per (b,h), Q(16384x16) x K^T(16x256) -> softmax ->
// x V(256x16). Block = 64 threads (2 waves, each a 16-row q tile); scores in
// LDS (2 x 16 x 256 f32 = 32 KB). grid = (512, 8, B).
// ---------------------------------------------------------------------------
__global__ void attn2_kernel(const float* __restrict__ q2, const float* __restrict__ kzn,
                             const float* __restrict__ kv, const float* __restrict__ temp,
                             float* __restrict__ oz) {
    const int Sh = 16384;
    int b = blockIdx.z, h = blockIdx.y;
    int lane = threadIdx.x & 31, wave = threadIdx.x >> 5;
    int row0 = (blockIdx.x * 2 + wave) * 16;
    __shared__ float Sb[2][16][256];
    float th = temp[h];
    const float* qbase = q2 + ((long)b * 128 + h * 16) * Sh;  // (c, n)
    const float* kb = kzn + ((long)b * 128 + h * 16) * 256;   // (c, tok)
    const float* vb = kv + ((long)b * 256 + 128 + h * 16) * 256;  // (c, tok)
    int am = lane & 15, akb = (lane >> 4) * 16;
    int bn = lane & 15, bkb = (lane >> 4) * 16;
    int mb = (lane >> 4) << 3;
    // phase A: scores S[m, tok] with K = hd(16) zero-padded to 32
    for (int nt = 0; nt < 16; ++nt) {
        v16bf af, bfv;
        v8f acc = {};
#pragma unroll
        for (int e = 0; e < 16; ++e) {
            int vg = e >> 1, hh = e & 1;
            int ka = akb + e;
            af[e] = (ka < 16) ? (bf16_t)qbase[(long)ka * Sh + row0 + am] : (bf16_t)0.0f;
        }
#pragma unroll
        for (int e = 0; e < 16; ++e) {
            int kg = bkb + e;
            bfv[e] = (kg < 16) ? (bf16_t)kb[(long)kg * 256 + nt * 16 + bn] : (bf16_t)0.0f;
        }
        acc = __builtin_amdgcn_wmma_f32_16x16x32_bf16(false, af, false, bfv,
                                                      (short)0, acc, false, false);
#pragma unroll
        for (int i = 0; i < 8; ++i) Sb[wave][mb + i][nt * 16 + (lane & 15)] = acc[i] * th;
    }
    __syncthreads();
    // phase B: softmax over 256 keys per row
    if (lane < 16) {
        float* row = Sb[wave][lane];
        float mx = -3.4e38f;
        for (int j = 0; j < 256; ++j) mx = fmaxf(mx, row[j]);
        float s = 0.f;
        for (int j = 0; j < 256; ++j) { float e = __expf(row[j] - mx); row[j] = e; s += e; }
        float is = 1.0f / s;
        for (int j = 0; j < 256; ++j) row[j] *= is;
    }
    __syncthreads();
    // phase C: O = P(16x256) x V(256x16), K = 256 in 8 WMMA steps
    v8f acc = {};
    for (int kt = 0; kt < 8; ++kt) {
        v16bf af, bfv;
#pragma unroll
        for (int e = 0; e < 16; ++e) {
            int vg = e >> 1, hh = e & 1;
            int ka = akb + e;
            af[e] = (bf16_t)Sb[wave][am][kt * 32 + ka];
        }
#pragma unroll
        for (int e = 0; e < 16; ++e) {
            int key = kt * 32 + bkb + e;
            bfv[e] = (bf16_t)vb[(long)bn * 256 + key];
        }
        acc = __builtin_amdgcn_wmma_f32_16x16x32_bf16(false, af, false, bfv,
                                                      (short)0, acc, false, false);
    }
    float* ob = oz + ((long)b * 128 + h * 16) * Sh;
#pragma unroll
    for (int i = 0; i < 8; ++i)
        ob[(long)(lane & 15) * Sh + row0 + mb + i] = acc[i];
}

// ===========================================================================
// Host orchestration
// ===========================================================================
extern "C" void kernel_launch(void* const* d_in, const int* in_sizes, int n_in,
                              void* d_out, int out_size, void* d_ws, size_t ws_size,
                              hipStream_t stream) {
    (void)in_sizes; (void)n_in; (void)out_size; (void)ws_size;
    enum {
        I_x = 0, I_prior, I_sp_norm_w, I_sp_norm_b, I_mb_exp_w, I_mb_exp_b, I_mb_dw_w,
        I_mb_dw_b, I_mb_se1_w, I_mb_se1_b, I_mb_se2_w, I_mb_se2_b, I_mb_proj_w,
        I_mb_proj_b, I_spa_int_w, I_spa_conv_w, I_spec_norm_w, I_spec_norm_b, I_qk_dw_w,
        I_qk_dw_b, I_qk_pw_w, I_qk_pw_b, I_qkvdw_w, I_v1_w, I_v1_b, I_v2_w, I_sid_w,
        I_sid_b, I_vd_dw_w, I_vd_dw_b, I_vd_pw_w, I_vd_pw_b, I_spec_temp, I_spec_po_dwt_w,
        I_spec_po_dwt_b, I_spec_po_pw_w, I_spec_po_pw_b, I_si1_w, I_si_ln_w, I_si_ln_b,
        I_si2_w, I_sp2p_dw_w, I_sp2p_dw_b, I_sp2p_pw_w, I_sp2p_pw_b, I_him_qkv_w,
        I_him_qkv_b, I_him_temp, I_him_po_dwt_w, I_him_po_dwt_b, I_him_po_pw_w,
        I_him_po_pw_b, I_add_ss_w, I_add_ss_b, I_add_ssp_w, I_add_ssp_b, I_ffn_ln_w,
        I_ffn_ln_b, I_ffn_in_w, I_ffn_dw_w, I_ffn_out_w
    };
    auto F = [&](int i) { return (const float*)d_in[i]; };
    const float* x = F(I_x);
    const float* prior = F(I_prior);

    constexpr int Bn = 2, HF = 256, WF = 256, H2 = 128, W2 = 128;
    constexpr long SF = 65536, SH = 16384;

    size_t off = 0;
    auto alloc = [&](size_t bytes) -> char* {
        off = (off + 255) & ~(size_t)255;
        char* p = (char*)d_ws + off;
        off += bytes;
        return p;
    };
    auto fbuf = [&](long n) { return (float*)alloc((size_t)n * 4); };
    auto bbuf = [&](long n) { return (bf16_t*)alloc((size_t)n * 2); };

    float* xs = fbuf(Bn * 64 * SF);
    float* xn = fbuf(Bn * 64 * SF);
    float* t = fbuf(Bn * 256 * SF);     // also reused as FFN t2 (pre-dw)
    float* tb = fbuf(Bn * 256 * SF);    // dw outputs (mb_dw / ffn_dw)
    float* sf = fbuf(Bn * 128 * SF);
    float* spint = fbuf(Bn * SF);
    float* qkd = fbuf(Bn * 64 * SH);
    float* qk1 = fbuf(Bn * 256 * SH);
    float* qk2 = fbuf(Bn * 256 * SH);
    float* v0 = fbuf(Bn * 128 * SF);
    float* vv = fbuf(Bn * 128 * SF);
    float* sid = fbuf(Bn * SH);
    float* vd0 = fbuf(Bn * 128 * SH);
    float* vdp = fbuf(Bn * 128 * SH);
    float* obuf = fbuf(Bn * 128 * SH);
    float* up = fbuf(Bn * 128 * SF);    // reused for both convT upsamples
    float* spec0 = fbuf(Bn * 64 * SF);
    float* q2d = fbuf(Bn * 128 * SH);
    float* q2p = fbuf(Bn * 128 * SH);
    float* kv = fbuf(Bn * 256 * 256);
    float* kzn = fbuf(Bn * 128 * 256);
    float* ozb = fbuf(Bn * 128 * SH);
    float* zfea = fbuf(Bn * 64 * SF);
    float* spabr = fbuf(Bn * 64 * SF);
    float* feass = fbuf(Bn * 64 * SF);
    float* fea = fbuf(Bn * 64 * SF);
    float* fnb = fbuf(Bn * 64 * SF);
    float* mean256 = fbuf(Bn * 256);
    float* sev = fbuf(Bn * 256);
    float* mean64 = fbuf(Bn * 64);
    float* giv = fbuf(Bn * 128);
    bf16_t* bfA = bbuf(Bn * 256 * SF);  // shared bf16 GEMM-input scratch

    bf16_t* w_mb_exp = bbuf(256 * 64);
    bf16_t* w_mb_proj = bbuf(128 * 256);
    bf16_t* w_qk_pw = bbuf(256 * 64);
    bf16_t* w_vd_pw = bbuf(128 * 128);
    bf16_t* w_spec_pw = bbuf(64 * 128);
    bf16_t* w_sp2p_pw = bbuf(128 * 128);
    bf16_t* w_him_qkv = bbuf(256 * 256);
    bf16_t* w_him_pw = bbuf(64 * 128);
    bf16_t* w_spa_conv = bbuf(64 * 128);
    bf16_t* w_add_ss = bbuf(64 * 64);
    bf16_t* w_add_ssp = bbuf(64 * 128);
    bf16_t* w_ffn_in = bbuf(256 * 64);
    bf16_t* w_ffn_out = bbuf(64 * 128);

    auto g1 = [&](long n) { return dim3((unsigned)((n + TPB - 1) / TPB)); };
    auto cvtW = [&](int idx, bf16_t* dst, long n) {
        cvt_bf16<<<g1(n), TPB, 0, stream>>>(F(idx), n, dst, n, n, n);
    };
    cvtW(I_mb_exp_w, w_mb_exp, 256 * 64);
    cvtW(I_mb_proj_w, w_mb_proj, 128 * 256);
    cvtW(I_qk_pw_w, w_qk_pw, 256 * 64);
    cvtW(I_vd_pw_w, w_vd_pw, 128 * 128);
    cvtW(I_spec_po_pw_w, w_spec_pw, 64 * 128);
    cvtW(I_sp2p_pw_w, w_sp2p_pw, 128 * 128);
    cvtW(I_him_qkv_w, w_him_qkv, 256 * 256);
    cvtW(I_him_po_pw_w, w_him_pw, 64 * 128);
    cvtW(I_spa_conv_w, w_spa_conv, 64 * 128);
    cvtW(I_add_ss_w, w_add_ss, 64 * 64);
    cvtW(I_add_ssp_w, w_add_ssp, 64 * 128);
    cvtW(I_ffn_in_w, w_ffn_in, 256 * 64);
    cvtW(I_ffn_out_w, w_ffn_out, 64 * 128);

    // ---- spatial branch ----
    ln_ch_kernel<<<g1(Bn * SF), TPB, 0, stream>>>(x, F(I_sp_norm_w), F(I_sp_norm_b), xs, 64, SF, Bn * SF);
    ln_ch_kernel<<<g1(Bn * SF), TPB, 0, stream>>>(x, F(I_spec_norm_w), F(I_spec_norm_b), xn, 64, SF, Bn * SF);

    cvt_bf16<<<g1(Bn * 64 * SF), TPB, 0, stream>>>(xs, 64 * SF, bfA, 64 * SF, 64 * SF, Bn * 64 * SF);
    wmma_gemm<1, true, false, true><<<dim3(SF / 64, 16, Bn), 128, 0, stream>>>(
        w_mb_exp, bfA, 64 * SF, F(I_mb_exp_b), nullptr, 0, t, 256 * SF, 256, 64, (int)SF, SF, 1);
    dw3x3<true><<<g1(Bn * 256 * SF), TPB, 0, stream>>>(t, F(I_mb_dw_w), F(I_mb_dw_b), tb, 256, HF, WF, Bn * 256 * SF);
    chan_mean<<<Bn * 256, 256, 0, stream>>>(tb, mean256, SF);
    se_gate<<<Bn, 256, 0, stream>>>(mean256, F(I_mb_se1_w), F(I_mb_se1_b), F(I_mb_se2_w), F(I_mb_se2_b), sev);
    scale_ch_bf16<<<g1(Bn * 256 * SF), TPB, 0, stream>>>(tb, sev, bfA, 256, SF, Bn * 256 * SF);
    wmma_gemm<0, true, false, true><<<dim3(SF / 64, 8, Bn), 128, 0, stream>>>(
        w_mb_proj, bfA, 256 * SF, F(I_mb_proj_b), nullptr, 0, sf, 128 * SF, 128, 256, (int)SF, SF, 1);
    dot_ch<<<g1(Bn * SF), TPB, 0, stream>>>(sf, F(I_spa_int_w), spint, 128, SF, Bn * SF);
    gelu_ip<<<g1(Bn * 128 * SF), TPB, 0, stream>>>(sf, Bn * 128 * SF);

    // ---- spectral branch: q/k/v ----
    dw2x2s2<<<g1(Bn * 64 * SH), TPB, 0, stream>>>(xn, F(I_qk_dw_w), F(I_qk_dw_b), qkd, 64, HF, WF, Bn * 64 * SH);
    cvt_bf16<<<g1(Bn * 64 * SH), TPB, 0, stream>>>(qkd, 64 * SH, bfA, 64 * SH, 64 * SH, Bn * 64 * SH);
    wmma_gemm<0, true, false, true><<<dim3(SH / 64, 16, Bn), 128, 0, stream>>>(
        w_qk_pw, bfA, 64 * SH, F(I_qk_pw_b), nullptr, 0, qk1, 256 * SH, 256, 64, (int)SH, SH, 1);
    grp2_3x3<<<g1(Bn * 256 * SH), TPB, 0, stream>>>(qk1, F(I_qkvdw_w), qk2, 256, H2, W2, Bn * 256 * SH);

    v1_kernel<<<g1(Bn * 128 * SF), TPB, 0, stream>>>(xn, F(I_v1_w), F(I_v1_b), v0, SF, Bn * 128 * SF);
    grp2_3x5<<<g1(Bn * 128 * SF), TPB, 0, stream>>>(v0, F(I_v2_w), vv, 128, HF, WF, Bn * 128 * SF);

    dw2x2s2<<<g1(Bn * SH), TPB, 0, stream>>>(spint, F(I_sid_w), F(I_sid_b), sid, 1, HF, WF, Bn * SH);
    mul_bcast<<<g1(Bn * 256 * SH), TPB, 0, stream>>>(qk2, sid, 256, SH, Bn * 256 * SH);      // q *= sid, k *= sid
    mul_bcast<<<g1(Bn * 128 * SF), TPB, 0, stream>>>(vv, spint, 128, SF, Bn * 128 * SF);     // v *= spint

    dw2x2s2<<<g1(Bn * 128 * SH), TPB, 0, stream>>>(vv, F(I_vd_dw_w), F(I_vd_dw_b), vd0, 128, HF, WF, Bn * 128 * SH);
    cvt_bf16<<<g1(Bn * 128 * SH), TPB, 0, stream>>>(vd0, 128 * SH, bfA, 128 * SH, 128 * SH, Bn * 128 * SH);
    wmma_gemm<0, true, false, true><<<dim3(SH / 64, 8, Bn), 128, 0, stream>>>(
        w_vd_pw, bfA, 128 * SH, F(I_vd_pw_b), nullptr, 0, vdp, 128 * SH, 128, 128, (int)SH, SH, 1);

    attn1_kernel<<<Bn * 8, 256, 0, stream>>>(qk2, vdp, F(I_spec_temp), obuf);
    dwT4x4s2<<<g1(Bn * 128 * SF), TPB, 0, stream>>>(obuf, F(I_spec_po_dwt_w), F(I_spec_po_dwt_b), up, 128, H2, W2, Bn * 128 * SF);
    cvt_bf16<<<g1(Bn * 128 * SF), TPB, 0, stream>>>(up, 128 * SF, bfA, 128 * SF, 128 * SF, Bn * 128 * SF);
    wmma_gemm<0, true, false, true><<<dim3(SF / 64, 4, Bn), 128, 0, stream>>>(
        w_spec_pw, bfA, 128 * SF, F(I_spec_po_pw_b), nullptr, 0, spec0, 64 * SF, 64, 128, (int)SF, SF, 1);

    // ---- gi gating of spatial features ----
    chan_mean<<<Bn * 64, 256, 0, stream>>>(spec0, mean64, SF);
    gi_gate<<<Bn, 128, 0, stream>>>(mean64, F(I_si1_w), F(I_si_ln_w), F(I_si_ln_b), F(I_si2_w), giv);
    scale_ch_bf16<<<g1(Bn * 128 * SF), TPB, 0, stream>>>(sf, giv, bfA, 128, SF, Bn * 128 * SF);
    wmma_gemm<0, false, true, true><<<dim3(SF / 64, 4, Bn), 128, 0, stream>>>(
        w_spa_conv, bfA, 128 * SF, nullptr, x, 64 * SF, spabr, 64 * SF, 64, 128, (int)SF, SF, 1);

    // ---- prior cross-attention ----
    dw2x2s2<<<g1(Bn * 128 * SH), TPB, 0, stream>>>(vv, F(I_sp2p_dw_w), F(I_sp2p_dw_b), q2d, 128, HF, WF, Bn * 128 * SH);
    cvt_bf16<<<g1(Bn * 128 * SH), TPB, 0, stream>>>(q2d, 128 * SH, bfA, 128 * SH, 128 * SH, Bn * 128 * SH);
    wmma_gemm<0, true, false, true><<<dim3(SH / 64, 8, Bn), 128, 0, stream>>>(
        w_sp2p_pw, bfA, 128 * SH, F(I_sp2p_pw_b), nullptr, 0, q2p, 128 * SH, 128, 128, (int)SH, SH, 1);

    cvt_bf16<<<g1(Bn * 256 * 256), TPB, 0, stream>>>(prior, 256 * 256, bfA, 256 * 256, 256 * 256, Bn * 256 * 256);
    // kv[e, tok] = W[e,:] . prior[tok,:]: B element (k,n) at n*256 + k
    wmma_gemm<0, true, false, false><<<dim3(4, 16, Bn), 128, 0, stream>>>(
        w_him_qkv, bfA, 256 * 256, F(I_him_qkv_b), nullptr, 0, kv, 256 * 256, 256, 256, 256, 1, 256);
    kz_norm<<<g1(Bn * 8 * 256), TPB, 0, stream>>>(kv, kzn, Bn * 8 * 256);
    attn2_kernel<<<dim3(512, 8, Bn), 64, 0, stream>>>(q2p, kzn, kv, F(I_him_temp), ozb);
    dwT4x4s2<<<g1(Bn * 128 * SF), TPB, 0, stream>>>(ozb, F(I_him_po_dwt_w), F(I_him_po_dwt_b), up, 128, H2, W2, Bn * 128 * SF);
    cvt_bf16<<<g1(Bn * 128 * SF), TPB, 0, stream>>>(up, 128 * SF, bfA, 128 * SF, 128 * SF, Bn * 128 * SF);
    wmma_gemm<0, true, false, true><<<dim3(SF / 64, 4, Bn), 128, 0, stream>>>(
        w_him_pw, bfA, 128 * SF, F(I_him_po_pw_b), nullptr, 0, zfea, 64 * SF, 64, 128, (int)SF, SF, 1);

    // ---- fusion ----
    // spa_br already = x + conv(sf_gated); spectral residual = x + spec0
    add3_bf16<<<g1(Bn * 64 * SF), TPB, 0, stream>>>(spabr, x, spec0, bfA, Bn * 64 * SF);
    wmma_gemm<0, true, false, true><<<dim3(SF / 64, 4, Bn), 128, 0, stream>>>(
        w_add_ss, bfA, 64 * SF, F(I_add_ss_b), nullptr, 0, feass, 64 * SF, 64, 64, (int)SF, SF, 1);
    // concat(fea_ss, z_fea) as bf16 (B,128,SF)
    cvt_bf16<<<g1(Bn * 64 * SF), TPB, 0, stream>>>(feass, 64 * SF, bfA, 128 * SF, 64 * SF, Bn * 64 * SF);
    cvt_bf16<<<g1(Bn * 64 * SF), TPB, 0, stream>>>(zfea, 64 * SF, bfA + 64 * SF, 128 * SF, 64 * SF, Bn * 64 * SF);
    wmma_gemm<0, true, false, true><<<dim3(SF / 64, 4, Bn), 128, 0, stream>>>(
        w_add_ssp, bfA, 128 * SF, F(I_add_ssp_b), nullptr, 0, fea, 64 * SF, 64, 128, (int)SF, SF, 1);

    // ---- FFN ----
    ln_ch_kernel<<<g1(Bn * SF), TPB, 0, stream>>>(fea, F(I_ffn_ln_w), F(I_ffn_ln_b), fnb, 64, SF, Bn * SF);
    cvt_bf16<<<g1(Bn * 64 * SF), TPB, 0, stream>>>(fnb, 64 * SF, bfA, 64 * SF, 64 * SF, Bn * 64 * SF);
    wmma_gemm<0, false, false, true><<<dim3(SF / 64, 16, Bn), 128, 0, stream>>>(
        w_ffn_in, bfA, 64 * SF, nullptr, nullptr, 0, t, 256 * SF, 256, 64, (int)SF, SF, 1);
    dw3x3<false><<<g1(Bn * 256 * SF), TPB, 0, stream>>>(t, F(I_ffn_dw_w), nullptr, tb, 256, HF, WF, Bn * 256 * SF);
    gate_fuse<<<g1(Bn * 128 * SF), TPB, 0, stream>>>(tb, bfA, Bn * 128 * SF);
    wmma_gemm<0, false, true, true><<<dim3(SF / 64, 4, Bn), 128, 0, stream>>>(
        w_ffn_out, bfA, 128 * SF, nullptr, fea, 64 * SF, (float*)d_out, 64 * SF, 64, 128, (int)SF, SF, 1);
}
